// AdditiveAttention_30992484008208
// MI455X (gfx1250) — matrix operand order repaired and checked by reading, not compiled, not run on hardware
//
#include <hip/hip_runtime.h>

// ---------------------------------------------------------------------------
// Problem constants (reference: B=32, S=4096, D=512)
// ---------------------------------------------------------------------------
#define BATCH 32
#define SEQ   4096
#define DIM   512
#define BS    (BATCH * SEQ)          // 131072 flattened rows
#define CTX_ELEMS ((size_t)BS * DIM) // context output elements

// WMMA vector types (probe-confirmed signatures)
typedef __attribute__((ext_vector_type(16))) __bf16 v16bf;
typedef __attribute__((ext_vector_type(8)))  float  v8f;

union FragBF16 {
    uint4 q[2];   // 32 bytes = 8 VGPRs = 16 bf16
    v16bf v;
};

// float -> bf16 with round-to-nearest-even, as raw ushort (avoids __bf16 scalar math)
__device__ __forceinline__ unsigned short f2bf(float f) {
    unsigned int u = __float_as_uint(f);
    u = (u + 0x7FFFu + ((u >> 16) & 1u)) >> 16;
    return (unsigned short)u;
}

__device__ __forceinline__ float fast_tanh(float x) {
    // tanh(x) = 1 - 2/(exp(2x)+1)  -> single v_exp_f32 + rcp path
    float e = __expf(2.0f * x);
    return 1.0f - 2.0f / (e + 1.0f);
}

// ---------------------------------------------------------------------------
// Kernel 0: f32 -> packed bf16 (weights staged once per launch into d_ws)
// ---------------------------------------------------------------------------
__global__ void cvt_f32_to_bf16(const float* __restrict__ src,
                                unsigned short* __restrict__ dst, int n4) {
    int i = blockIdx.x * blockDim.x + threadIdx.x;
    if (i < n4) {
        float4 f = ((const float4*)src)[i];
        uint2 p;
        p.x = (unsigned int)f2bf(f.x) | ((unsigned int)f2bf(f.y) << 16);
        p.y = (unsigned int)f2bf(f.z) | ((unsigned int)f2bf(f.w) << 16);
        ((uint2*)dst)[i] = p;
    }
}

// ---------------------------------------------------------------------------
// Kernel 1: fused scores = V . tanh(Q*Wq^T + V*Wk^T + bq + bk) + vb
// Block: 64 threads (2 waves), 32 rows per block, grid = BS/32 = 4096.
// LDS: 32x512 bf16 query tile + 32x512 bf16 value tile = 64 KB.
// A-fragments (16 k-steps x 2 matrices = 256 VGPRs) are held in registers
// for the whole e-loop; __launch_bounds__(64,1) raises the VGPR budget so
// they do NOT spill to scratch.
// ---------------------------------------------------------------------------
__global__ void __launch_bounds__(64, 1)
scores_kernel(const float* __restrict__ query,
              const float* __restrict__ value,
              const unsigned short* __restrict__ wq_bf, // [512][512] bf16, row = e
              const unsigned short* __restrict__ wk_bf,
              const float* __restrict__ bq,
              const float* __restrict__ bk,
              const float* __restrict__ vw,
              const float* __restrict__ vb,
              float* __restrict__ scores) {
    __shared__ unsigned short Aq[32 * DIM]; // 32 KB
    __shared__ unsigned short Av[32 * DIM]; // 32 KB

    const int tid = threadIdx.x;                 // 0..63
    const long rowBase = (long)blockIdx.x * 32;  // first flattened row

    // ---- Phase A: stage 32 rows of query & value into LDS as bf16 ----------
    // 32 rows * 128 float4 chunks = 4096 chunks, 64 threads -> 64 each.
    for (int c = tid; c < 32 * (DIM / 4); c += 64) {
        float4 fq = ((const float4*)query)[rowBase * (DIM / 4) + c];
        float4 fv = ((const float4*)value)[rowBase * (DIM / 4) + c];
        uint2 pq, pv;
        pq.x = (unsigned int)f2bf(fq.x) | ((unsigned int)f2bf(fq.y) << 16);
        pq.y = (unsigned int)f2bf(fq.z) | ((unsigned int)f2bf(fq.w) << 16);
        pv.x = (unsigned int)f2bf(fv.x) | ((unsigned int)f2bf(fv.y) << 16);
        pv.y = (unsigned int)f2bf(fv.z) | ((unsigned int)f2bf(fv.w) << 16);
        ((uint2*)Aq)[c] = pq;
        ((uint2*)Av)[c] = pv;
    }
    __syncthreads();

    // ---- Phase B: per-wave 16x16 WMMA tiles over all 32 e-tiles ------------
    const int lane = tid & 31;
    const int wv   = tid >> 5;        // wave id: 0..1 -> rows wv*16..wv*16+15
    const int g    = lane >> 4;       // half-wave selector
    const int mA   = lane & 15;       // A-matrix row within tile (lanes 0-15 == 16-31)
    const int n    = lane & 15;       // B/C column

    const unsigned short* aq_row = &Aq[(wv * 16 + mA) * DIM];
    const unsigned short* av_row = &Av[(wv * 16 + mA) * DIM];

    // Load ALL A-fragments into registers once (reused for all 32 e-tiles).
    // ISA 16-bit A 16x32 layout: lane g*16+m, VGPRs 0-3 <- K=g*8..g*8+7,
    // VGPRs 4-7 <- K=16+g*8..16+g*8+7  == two 16-byte LDS loads per fragment.
    FragBF16 aq[16], av[16];
#pragma unroll
    for (int ks = 0; ks < 16; ++ks) {
        const int kb = ks * 32;
        aq[ks].q[0] = *(const uint4*)(aq_row + kb + g * 8);
        aq[ks].q[1] = *(const uint4*)(aq_row + kb + 16 + g * 8);
        av[ks].q[0] = *(const uint4*)(av_row + kb + g * 8);
        av[ks].q[1] = *(const uint4*)(av_row + kb + 16 + g * 8);
    }

    float sacc[8];
#pragma unroll
    for (int j = 0; j < 8; ++j) sacc[j] = 0.0f;

    for (int et = 0; et < 32; ++et) {
        const int e0 = et * 16;
        v8f acc = {};

        // B fragment rows (ISA B 32x16 layout): lane g*16+n holds K=g*8..g*8+7
        // and K=16+g*8..16+g*8+7 of weight row e0+n  == two 16-byte global loads.
        const unsigned short* wqr = wq_bf + (size_t)(e0 + n) * DIM + g * 8;
        const unsigned short* wkr = wk_bf + (size_t)(e0 + n) * DIM + g * 8;

        // Prefetch next e-tile's weight rows into cache (global_prefetch_b8).
        if (et < 31) {
            __builtin_prefetch(wqr + 16 * DIM, 0, 1);
            __builtin_prefetch(wkr + 16 * DIM, 0, 1);
        }

        // K-loop over query projection: 16 steps of K=32
#pragma unroll
        for (int ks = 0; ks < 16; ++ks) {
            const int kb = ks * 32;
            FragBF16 b;
            b.q[0] = *(const uint4*)(wqr + kb);
            b.q[1] = *(const uint4*)(wqr + kb + 16);
            acc = __builtin_amdgcn_wmma_f32_16x16x32_bf16(
                false, aq[ks].v, false, b.v, (short)0, acc, false, false);
        }
        // K-loop over key projection (from value)
#pragma unroll
        for (int ks = 0; ks < 16; ++ks) {
            const int kb = ks * 32;
            FragBF16 b;
            b.q[0] = *(const uint4*)(wkr + kb);
            b.q[1] = *(const uint4*)(wkr + kb + 16);
            acc = __builtin_amdgcn_wmma_f32_16x16x32_bf16(
                false, av[ks].v, false, b.v, (short)0, acc, false, false);
        }

        // Fused epilogue: + bias, tanh, * V_w[e], accumulate over e
        const float bias = bq[e0 + n] + bk[e0 + n];
        const float w    = vw[e0 + n];
#pragma unroll
        for (int j = 0; j < 8; ++j) {
            sacc[j] += fast_tanh(acc[j] + bias) * w;
        }
    }

    // Reduce over the 16 columns (n) held by the 16 lanes of each half-wave.
#pragma unroll
    for (int j = 0; j < 8; ++j) {
        sacc[j] += __shfl_xor(sacc[j], 1, 32);
        sacc[j] += __shfl_xor(sacc[j], 2, 32);
        sacc[j] += __shfl_xor(sacc[j], 4, 32);
        sacc[j] += __shfl_xor(sacc[j], 8, 32);
    }

    if (n == 0) {
        const float v_b = vb[0];
#pragma unroll
        for (int j = 0; j < 8; ++j) {
            // C layout: lane group g holds rows M = j + 8*g
            scores[rowBase + wv * 16 + (j + 8 * g)] = sacc[j] + v_b;
        }
    }
}

// ---------------------------------------------------------------------------
// Kernel 2: row softmax over S=4096 per batch (32 blocks x 256 threads)
// ---------------------------------------------------------------------------
__global__ void softmax_kernel(const float* __restrict__ scores,
                               float* __restrict__ attn) {
    __shared__ float red[256];
    const int b = blockIdx.x;
    const int t = threadIdx.x;
    const float* s = scores + (size_t)b * SEQ;

    float mx = -3.0e38f;
    for (int i = t; i < SEQ; i += 256) mx = fmaxf(mx, s[i]);
    red[t] = mx;
    __syncthreads();
    for (int o = 128; o > 0; o >>= 1) {
        if (t < o) red[t] = fmaxf(red[t], red[t + o]);
        __syncthreads();
    }
    const float M = red[0];
    __syncthreads();

    float sum = 0.0f;
    for (int i = t; i < SEQ; i += 256) sum += __expf(s[i] - M);
    red[t] = sum;
    __syncthreads();
    for (int o = 128; o > 0; o >>= 1) {
        if (t < o) red[t] += red[t + o];
        __syncthreads();
    }
    const float inv = 1.0f / red[0];

    for (int i = t; i < SEQ; i += 256) {
        attn[(size_t)b * SEQ + i] = __expf(s[i] - M) * inv;
    }
}

// ---------------------------------------------------------------------------
// Kernel 3: context[b,s,:] = attn[b,s] * value[b,s,:]  (bandwidth bound)
// ---------------------------------------------------------------------------
__global__ void context_kernel(const float* __restrict__ value,
                               const float* __restrict__ attn,
                               float* __restrict__ ctx, long n4) {
    long i = (long)blockIdx.x * blockDim.x + threadIdx.x;
    const long stride = (long)gridDim.x * blockDim.x;
    for (; i < n4; i += stride) {
        const float a = attn[i >> 7];           // 128 float4 chunks per row of D=512
        float4 v = ((const float4*)value)[i];
        v.x *= a; v.y *= a; v.z *= a; v.w *= a;
        ((float4*)ctx)[i] = v;
    }
}

// ---------------------------------------------------------------------------
// Launch
// ---------------------------------------------------------------------------
extern "C" void kernel_launch(void* const* d_in, const int* in_sizes, int n_in,
                              void* d_out, int out_size, void* d_ws, size_t ws_size,
                              hipStream_t stream) {
    const float* query = (const float*)d_in[0];
    const float* value = (const float*)d_in[1];
    const float* Wq_w  = (const float*)d_in[2];
    const float* Wq_b  = (const float*)d_in[3];
    const float* Wk_w  = (const float*)d_in[4];
    const float* Wk_b  = (const float*)d_in[5];
    const float* V_w   = (const float*)d_in[6];
    const float* V_b   = (const float*)d_in[7];

    float* ctx  = (float*)d_out;                 // [B,S,D]
    float* attn = ctx + CTX_ELEMS;               // [B,S]

    // Workspace: bf16 Wq (512KB) | bf16 Wk (512KB) | scores f32 (512KB)
    unsigned short* wq_bf = (unsigned short*)d_ws;
    unsigned short* wk_bf = wq_bf + (size_t)DIM * DIM;
    float* scores = (float*)((char*)d_ws + 2ull * DIM * DIM * sizeof(unsigned short));

    // 0) weights f32 -> bf16 (DIM*DIM/4 = 65536 float4 chunks each)
    cvt_f32_to_bf16<<<256, 256, 0, stream>>>(Wq_w, wq_bf, DIM * DIM / 4);
    cvt_f32_to_bf16<<<256, 256, 0, stream>>>(Wk_w, wk_bf, DIM * DIM / 4);

    // 1) fused WMMA scores: 4096 blocks x 64 threads (2 waves, 32 rows each)
    scores_kernel<<<BS / 32, 64, 0, stream>>>(query, value, wq_bf, wk_bf,
                                              Wq_b, Wk_b, V_w, V_b, scores);

    // 2) softmax per batch row
    softmax_kernel<<<BATCH, 256, 0, stream>>>(scores, attn);

    // 3) context = attn * value
    context_kernel<<<4096, 256, 0, stream>>>(value, attn, ctx,
                                             (long)(CTX_ELEMS / 4));
}
